// GNNEncoder_73495480369260
// MI455X (gfx1250) — matrix operand order repaired and checked by reading, not compiled, not run on hardware
//
#include <hip/hip_runtime.h>

#define N_NODES 50000
#define N_EDGES 800000
#define D_FEAT  128   // D_IN == D_HID == 128

typedef __attribute__((ext_vector_type(16))) _Float16 v16h;
typedef __attribute__((ext_vector_type(8)))  float    v8f;

// ---------------------------------------------------------------------------
// Zero a float buffer.
// ---------------------------------------------------------------------------
__global__ __launch_bounds__(256) void zero_f32(float* __restrict__ p, long long n) {
    long long i = (long long)blockIdx.x * blockDim.x + threadIdx.x;
    if (i < n) p[i] = 0.0f;
}

// ---------------------------------------------------------------------------
// Pack stacked weights [Wl ; Wr] (K=256 rows, Dout cols, f32 row-major) into
// WMMA-B fragment-ready f16 layout:  dst[((ntile*8 + ktile)*32 + lane)*16 + e]
// where lane -> (grp = lane>>4, m = lane&15) and the fragment element e maps
// to B[k = ktile*32 + (e<8 ? grp*8+e : 16+grp*8+e-8)][n = ntile*16 + m]  (ISA 7.12.2, 16-bit B).
// ---------------------------------------------------------------------------
__global__ __launch_bounds__(256) void pack_w(
    const float* __restrict__ Wl, const float* __restrict__ Wr,
    _Float16* __restrict__ dst, int Dout)
{
    int tid = blockIdx.x * blockDim.x + threadIdx.x;
    int total = (Dout / 16) * 8 * 32 * 16;
    if (tid >= total) return;
    const int e     = tid & 15;
    const int lane  = (tid >> 4) & 31;
    const int ktile = (tid >> 9) & 7;
    const int ntile = tid >> 12;
    const int grp = lane >> 4;
    const int m   = lane & 15;
    const int k = ktile * 32 + ((e < 8) ? (grp * 8 + e) : (16 + grp * 8 + (e - 8)));     // 0..255
    const int n = ntile * 16 + m;
    const float v = (k < D_FEAT) ? Wl[k * Dout + n]
                                 : Wr[(k - D_FEAT) * Dout + n];
    dst[tid] = (_Float16)v;
}

// ---------------------------------------------------------------------------
// Edge scatter: agg[dst] += feat[src], deg[dst] += 1 (optionally).
// One lane handles one (edge, 4-float chunk). 128 floats per edge = 32 lanes.
// ---------------------------------------------------------------------------
__global__ __launch_bounds__(256) void scatter_edges(
    const float* __restrict__ feat,
    const int*   __restrict__ ei,      // [2, N_EDGES] row-major: src then dst
    float* __restrict__ agg,
    float* __restrict__ deg,
    int addDeg)
{
    long long tid = (long long)blockIdx.x * blockDim.x + threadIdx.x;
    const long long total = (long long)N_EDGES * 32;
    if (tid >= total) return;
    const int e    = (int)(tid >> 5);
    const int lane = (int)(tid & 31);
    const int s = ei[e];
    const int d = ei[N_EDGES + e];

    const float4 val = *(const float4*)(feat + (size_t)s * D_FEAT + lane * 4);
    float* a = agg + (size_t)d * D_FEAT + lane * 4;
    atomicAdd(a + 0, val.x);
    atomicAdd(a + 1, val.y);
    atomicAdd(a + 2, val.z);
    atomicAdd(a + 3, val.w);
    if (addDeg && lane == 0) atomicAdd(deg + d, 1.0f);
}

// ---------------------------------------------------------------------------
// Fused SAGE layer:  out = act( (agg/deg) @ Wl + xin @ Wr + b )
// Viewed as [N, 256] @ [256, Dout]. ONE WAVE computes a full 16 x Dout row
// block: per k-tile the A fragment is built once (two contiguous 8-float runs
// per lane = ISA 7.12.2 16-bit A layout) and reused by NT = Dout/16
// back-to-back WMMAs against pre-packed B fragments. No LDS, no barriers.
// ---------------------------------------------------------------------------
template<int NT, bool RELU>
__global__ __launch_bounds__(32) void sage_wmma(
    const float* __restrict__ agg,
    const float* __restrict__ deg,
    const float* __restrict__ xin,
    const _Float16* __restrict__ Wpack,
    const float* __restrict__ bias,
    float* __restrict__ out)
{
    constexpr int Dout = NT * 16;
    const int lane = threadIdx.x;          // 0..31
    const int m    = lane & 15;
    const int grp  = lane >> 4;
    const int row0 = blockIdx.x * 16;      // 3125 row blocks, exact

    const int   row    = row0 + m;
    const float invdeg = 1.0f / fmaxf(deg[row], 1.0f);

    // Per-lane base of the packed B fragments.
    const _Float16* wp = Wpack + (size_t)lane * 16;

    v8f acc[NT] = {};

    #pragma unroll
    for (int ktile = 0; ktile < 8; ++ktile) {
        const int k0 = ktile * 32;
        const float* src;
        float scale;
        int kk;
        if (k0 < D_FEAT) { src = agg; scale = invdeg; kk = k0; }
        else             { src = xin; scale = 1.0f;   kk = k0 - D_FEAT; }

        // Two contiguous 8-float runs: k = kk+grp*8 .. +7 and kk+16+grp*8 .. +7
        const float* pr = src + (size_t)row * D_FEAT + kk + grp * 8;
        const float4 q0 = *(const float4*)(pr);
        const float4 q1 = *(const float4*)(pr + 4);
        const float4 q2 = *(const float4*)(pr + 16);
        const float4 q3 = *(const float4*)(pr + 20);

        v16h af;
        af[0]  = (_Float16)(q0.x * scale);
        af[1]  = (_Float16)(q0.y * scale);
        af[2]  = (_Float16)(q0.z * scale);
        af[3]  = (_Float16)(q0.w * scale);
        af[4]  = (_Float16)(q1.x * scale);
        af[5]  = (_Float16)(q1.y * scale);
        af[6]  = (_Float16)(q1.z * scale);
        af[7]  = (_Float16)(q1.w * scale);
        af[8]  = (_Float16)(q2.x * scale);
        af[9]  = (_Float16)(q2.y * scale);
        af[10] = (_Float16)(q2.z * scale);
        af[11] = (_Float16)(q2.w * scale);
        af[12] = (_Float16)(q3.x * scale);
        af[13] = (_Float16)(q3.y * scale);
        af[14] = (_Float16)(q3.z * scale);
        af[15] = (_Float16)(q3.w * scale);

        // One A fragment feeds NT WMMAs (distinct accumulators -> no hazards).
        #pragma unroll
        for (int nt = 0; nt < NT; ++nt) {
            const v16h bf = *reinterpret_cast<const v16h*>(
                wp + ((size_t)nt * 8 + ktile) * 512);
            acc[nt] = __builtin_amdgcn_wmma_f32_16x16x32_f16(
                /*neg_a=*/false, af, /*neg_b=*/false, bf,
                /*c_mod=*/(short)0, acc[nt], /*reuse_a=*/false, /*reuse_b=*/false);
        }
    }

    // ---- epilogue: C/D layout -> row = v + 8*grp, col = lane%16 ----
    const int mg = grp * 8;
    #pragma unroll
    for (int nt = 0; nt < NT; ++nt) {
        const float bv = bias[nt * 16 + m];
        #pragma unroll
        for (int v = 0; v < 8; ++v) {
            float r = acc[nt][v] + bv;
            if (RELU) r = fmaxf(r, 0.0f);
            out[(size_t)(row0 + mg + v) * Dout + nt * 16 + m] = r;
        }
    }
}

// ---------------------------------------------------------------------------
// Host-side launch.
// Inputs: x, edge_index, W1_l, W1_r, b1, W2_l, W2_r, b2
// ---------------------------------------------------------------------------
extern "C" void kernel_launch(void* const* d_in, const int* in_sizes, int n_in,
                              void* d_out, int out_size, void* d_ws, size_t ws_size,
                              hipStream_t stream) {
    const float* x   = (const float*)d_in[0];
    const int*   ei  = (const int*)  d_in[1];
    const float* W1l = (const float*)d_in[2];
    const float* W1r = (const float*)d_in[3];
    const float* b1  = (const float*)d_in[4];
    const float* W2l = (const float*)d_in[5];
    const float* W2r = (const float*)d_in[6];
    const float* b2  = (const float*)d_in[7];
    float* out = (float*)d_out;

    // Workspace (floats): deg[50048] | agg[N*128] | h[N*128] | wp1 | wp2 (f16)
    float* wsF = (float*)d_ws;
    float* deg = wsF;
    float* agg = wsF + 50048;
    float* h   = agg + (size_t)N_NODES * D_FEAT;
    _Float16* wp1 = (_Float16*)(h + (size_t)N_NODES * D_FEAT);   // 32768 f16
    _Float16* wp2 = wp1 + 32768;                                  // 16384 f16

    const long long aggN  = (long long)N_NODES * D_FEAT;
    const long long zeroN = 50048 + aggN;             // deg + agg contiguous
    const long long scatT = (long long)N_EDGES * 32;

    // ---- weight pre-pack (fragment-ready f16) ----
    pack_w<<<(32768 + 255) / 256, 256, 0, stream>>>(W1l, W1r, wp1, 128);
    pack_w<<<(16384 + 255) / 256, 256, 0, stream>>>(W2l, W2r, wp2, 64);

    // ---- Layer 1 ----
    zero_f32<<<(unsigned)((zeroN + 255) / 256), 256, 0, stream>>>(wsF, zeroN);
    scatter_edges<<<(unsigned)((scatT + 255) / 256), 256, 0, stream>>>(
        x, ei, agg, deg, /*addDeg=*/1);
    sage_wmma<8, true><<<N_NODES / 16, 32, 0, stream>>>(
        agg, deg, x, wp1, b1, h);

    // ---- Layer 2 (degrees unchanged; re-zero agg only) ----
    zero_f32<<<(unsigned)((aggN + 255) / 256), 256, 0, stream>>>(agg, aggN);
    scatter_edges<<<(unsigned)((scatT + 255) / 256), 256, 0, stream>>>(
        h, ei, agg, deg, /*addDeg=*/0);
    sage_wmma<4, false><<<N_NODES / 16, 32, 0, stream>>>(
        agg, deg, h, wp2, b2, out);
}
